// glimpse_network_38405597561374
// MI455X (gfx1250) — matrix operand order repaired and checked by reading, not compiled, not run on hardware
//
#include <hip/hip_runtime.h>
#include <stdint.h>

typedef _Float16 h16 __attribute__((ext_vector_type(16)));
typedef _Float16 h8  __attribute__((ext_vector_type(8)));
typedef float    f8  __attribute__((ext_vector_type(8)));

// d_in flattened in setup_inputs() insertion order:
//  0:x 1:l_t_prev 2:size_t_prev
//  towers s=2,4,8,16,32: per layer {bw1,bg1,bb1,bw2,bg2,bb2,cw,cb,dg,db}, then ew,eb
//  bases: 3,15,37,69,111 ; ew = base+10n ; then 163:w1 164:b1 165:w2 166:b2
//  167:what_w 168:what_b 169:where_w 170:where_b 171:size_w 172:size_b
struct Params { const float* p[173]; };

#define DTOT 768
#define HG   512

// ---------------- pack B^T (768 x 768, n-major) + summed bias ----------------
__global__ __launch_bounds__(256) void pack_b_kernel(Params prm, _Float16* Bt, float* bias) {
    int idx = blockIdx.x * 256 + threadIdx.x;
    if (idx < DTOT * DTOT) {
        int n = idx / DTOT, k = idx - n * DTOT;
        float v;
        if (k < HG)            v = prm.p[167][n * HG + k];               // what_w (768,512)
        else if (k < HG + 128) v = prm.p[169][n * 128 + (k - HG)];       // where_w (768,128)
        else                   v = prm.p[171][n * 128 + (k - HG - 128)]; // size_w (768,128)
        Bt[idx] = (_Float16)v;
    }
    if (idx < DTOT) bias[idx] = prm.p[168][idx] + prm.p[170][idx] + prm.p[172][idx];
}

// ---------------- per-sample l_out / s_out MLPs + size argmax ----------------
__global__ __launch_bounds__(256) void prep_kernel(Params prm, int* d_sidx, _Float16* Arow) {
    int b = blockIdx.x, t = threadIdx.x;
    const float* l  = prm.p[1];
    const float* sp = prm.p[2];
    if (t < 128) {  // l_out = relu(l @ w1^T + b1)
        const float* w1 = prm.p[163]; const float* b1 = prm.p[164];
        float v = l[b * 2 + 0] * w1[t * 2 + 0] + l[b * 2 + 1] * w1[t * 2 + 1] + b1[t];
        Arow[b * DTOT + HG + t] = (_Float16)fmaxf(v, 0.f);
    } else if (t < 256) {  // s_out = relu(size_prev @ w2^T + b2)
        int k = t - 128;
        const float* w2 = prm.p[165]; const float* b2 = prm.p[166];
        float acc = b2[k];
        for (int i = 0; i < 5; ++i) acc += sp[b * 5 + i] * w2[k * 5 + i];
        Arow[b * DTOT + HG + 128 + k] = (_Float16)fmaxf(acc, 0.f);
    }
    if (t == 0) {  // first-occurrence argmax
        float best = sp[b * 5]; int bi = 0;
        for (int i = 1; i < 5; ++i) { float v = sp[b * 5 + i]; if (v > best) { best = v; bi = i; } }
        d_sidx[b] = bi;
    }
}

// ------- stable counting sort position: perm[pos[b]] = b, key desc size_idx -------
__global__ __launch_bounds__(1024) void sort_kernel(const int* d_sidx, int* d_pos, int B) {
    __shared__ int key[1024];
    int t = threadIdx.x;
    if (t < B) key[t] = d_sidx[t];
    __syncthreads();
    if (t < B) {
        int my = key[t], cnt = 0;
        for (int j = 0; j < B; ++j) {
            int kj = key[j];
            cnt += (kj > my) || (kj == my && j < t);
        }
        d_pos[t] = cnt;
    }
}

// ---------------- tower: one block per sample, LDS-resident ----------------
__device__ __forceinline__ void conv3x3_bn(const float* in, float* out, const float* w,
                                           const float* g, const float* bb,
                                           int C, int S, bool relu, int t) {
    int SS = S * S;
    for (int o = t; o < C * SS; o += 256) {
        int co = o / SS, rem = o - co * SS, yy = rem / S, xx = rem - yy * S;
        float acc = 0.f;
        for (int ci = 0; ci < C; ++ci) {
            const float* wv = w + (co * C + ci) * 9;
            const float* av = in + ci * SS;
            #pragma unroll
            for (int dy = 0; dy < 3; ++dy) {
                int y = yy + dy - 1;
                if ((unsigned)y >= (unsigned)S) continue;
                #pragma unroll
                for (int dx = 0; dx < 3; ++dx) {
                    int x = xx + dx - 1;
                    if ((unsigned)x >= (unsigned)S) continue;
                    acc += av[y * S + x] * wv[dy * 3 + dx];
                }
            }
        }
        float v = acc * g[co] + bb[co];
        out[o] = relu ? fmaxf(v, 0.f) : v;
    }
}

__global__ __launch_bounds__(256) void tower_kernel(Params prm, const int* d_sidx,
                                                    const int* d_pos, _Float16* Arow) {
    __shared__ float bufA[3072], bufB[3072], bufC[3072];
    const int planes[6] = {3, 12, 24, 48, 96, 192};
    const int tbase[5]  = {3, 15, 37, 69, 111};
    const int b = blockIdx.x, t = threadIdx.x;
    const int sidx = d_sidx[b];
    const int s = 2 << sidx;          // 2,4,8,16,32
    const int n = sidx + 1;
    const float* x = prm.p[0];
    const float* l = prm.p[1];
    // coords: trunc(0.5*((l+1)*128)), non-negative so C-cast == astype(int32)
    int c0 = (int)(0.5f * ((l[b * 2 + 0] + 1.0f) * 128.0f));
    int c1 = (int)(0.5f * ((l[b * 2 + 1] + 1.0f) * 128.0f));
    // extract zero-padded patch (C,s,s), center (c1,c0)
    int np = 3 * s * s;
    for (int o = t; o < np; o += 256) {
        int ch = o / (s * s), rem = o - ch * s * s, i = rem / s, j = rem - i * s;
        int r = c1 - s / 2 + i, c = c0 - s / 2 + j;
        float v = 0.f;
        if (r >= 0 && r < 128 && c >= 0 && c < 128) v = x[((b * 3 + ch) * 128 + r) * 128 + c];
        bufA[o] = v;
    }
    __syncthreads();
    int S = s;
    for (int L = 0; L < n; ++L) {
        const int Ci = planes[L], Co = planes[L + 1];
        const float* const* pp = prm.p + tbase[sidx] + 10 * L;
        const int SS = S * S;
        conv3x3_bn(bufA, bufB, pp[0], pp[1], pp[2], Ci, S, true, t);   // conv1+bn+relu
        __syncthreads();
        conv3x3_bn(bufB, bufC, pp[3], pp[4], pp[5], Ci, S, false, t);  // conv2+bn
        __syncthreads();
        for (int o = t; o < Ci * SS; o += 256) bufA[o] = fmaxf(bufC[o] + bufA[o], 0.f); // residual
        __syncthreads();
        const float *cw = pp[6], *cb = pp[7], *dg = pp[8], *db = pp[9];
        int S2 = S >> 1, SS2 = S2 * S2;
        for (int o = t; o < Co * SS2; o += 256) {  // 2x2 stride-2 conv + bias + bn + relu
            int co = o / SS2, rem = o - co * SS2, yy = rem / S2, xx = rem - yy * S2;
            float acc = cb[co];
            for (int ci = 0; ci < Ci; ++ci) {
                const float* a = bufA + ci * SS + (2 * yy) * S + 2 * xx;
                const float* w = cw + (co * Ci + ci) * 4;
                acc += a[0] * w[0] + a[1] * w[1] + a[S] * w[2] + a[S + 1] * w[3];
            }
            bufB[o] = fmaxf(acc * dg[co] + db[co], 0.f);
        }
        __syncthreads();
        for (int o = t; o < Co * SS2; o += 256) bufA[o] = bufB[o];
        __syncthreads();
        S = S2;
    }
    // FC planes[n] -> 512, relu; scatter row to pos[b] (implements perm)
    int Pn = planes[n];
    const float* ew = prm.p[tbase[sidx] + 10 * n];
    const float* eb = prm.p[tbase[sidx] + 10 * n + 1];
    long row = d_pos[b];
    for (int j = t; j < HG; j += 256) {
        float acc = eb[j];
        for (int k = 0; k < Pn; ++k) acc += bufA[k] * ew[j * Pn + k];
        Arow[row * DTOT + j] = (_Float16)fmaxf(acc, 0.f);
    }
}

// ---------------- fused WMMA GEMM with async-to-LDS B staging ----------------
// Block: 8 waves -> 128 M-rows x 64 N-cols. B tile (64 rows x 32 halves = 4KB)
// double-buffered in LDS via global_load_async_to_lds_b128 (ASYNCcnt).
__device__ __forceinline__ void async_b128_to_lds(uint32_t lds_off, const void* gaddr) {
    // VDST = LDS byte address VGPR, VADDR = 64-bit global address, no SADDR
    asm volatile("global_load_async_to_lds_b128 %0, %1, off"
                 :: "v"(lds_off), "v"(gaddr) : "memory");
}

__global__ __launch_bounds__(256) void gemm_kernel(const _Float16* __restrict__ A,
                                                   const _Float16* __restrict__ Bt,
                                                   const float* __restrict__ bias,
                                                   float* __restrict__ out) {
    __shared__ __attribute__((aligned(128))) _Float16 btile[2][64 * 32]; // 2 x 4KB
    const int tid  = threadIdx.x;
    const int w    = tid >> 5, lane = tid & 31;
    const int bm   = blockIdx.x / 12, bn = blockIdx.x - bm * 12;
    const int mBase = bm * 128 + w * 16;
    const int nBase = bn * 64;
    const int half16 = lane >> 4;
    const int aoff   = half16 * 16;   // A: 16-half K split per half-wave
    const int boff   = half16 * 16;   // B: 16-half K split per half-wave
    const _Float16* arow = A + (long)(mBase + (lane & 15)) * DTOT;

    // per-thread async copy assignment: 256 threads x 16B = 128 rows x 32B
    const int nrow = tid >> 1, chunk = tid & 1;
    const _Float16* gsrc = Bt + (long)(nBase + nrow) * DTOT + chunk * 8;
    // low 32 bits of a flat shared pointer = workgroup-relative LDS byte address
    const uint32_t ldsbase = (uint32_t)(uintptr_t)(void*)&btile[0][0];
    const uint32_t ldst    = ldsbase + (uint32_t)(nrow * 64 + chunk * 16);

    f8 acc0 = {}, acc1 = {}, acc2 = {}, acc3 = {};
    async_b128_to_lds(ldst, gsrc);                       // preload kk = 0 into buf 0
    #pragma unroll 1
    for (int step = 0; step < DTOT / 32; ++step) {
        const int cur = step & 1;
        if (step + 1 < DTOT / 32) {
            async_b128_to_lds(ldst + (cur ^ 1) * 4096, gsrc + (step + 1) * 32);
            asm volatile("s_wait_asynccnt 0x1" ::: "memory"); // prev done, next in flight
        } else {
            asm volatile("s_wait_asynccnt 0x0" ::: "memory");
        }
        __syncthreads();                                 // cross-wave LDS visibility
        const int kk = step * 32;
        h8 alo = *(const h8*)(arow + kk + aoff);
        h8 ahi = *(const h8*)(arow + kk + 8 + aoff);
        h16 a = __builtin_shufflevector(alo, ahi, 0, 1, 2, 3, 4, 5, 6, 7,
                                        8, 9, 10, 11, 12, 13, 14, 15);
        const _Float16* lb = &btile[cur][(lane & 15) * 32 + boff];
        h16 b0 = *(const h16*)(lb);
        h16 b1 = *(const h16*)(lb + 16 * 32);
        h16 b2 = *(const h16*)(lb + 32 * 32);
        h16 b3 = *(const h16*)(lb + 48 * 32);
        acc0 = __builtin_amdgcn_wmma_f32_16x16x32_f16(false, a, false, b0, (short)0, acc0, false, false);
        acc1 = __builtin_amdgcn_wmma_f32_16x16x32_f16(false, a, false, b1, (short)0, acc1, false, false);
        acc2 = __builtin_amdgcn_wmma_f32_16x16x32_f16(false, a, false, b2, (short)0, acc2, false, false);
        acc3 = __builtin_amdgcn_wmma_f32_16x16x32_f16(false, a, false, b3, (short)0, acc3, false, false);
        __syncthreads();                                 // protect buffer reuse
    }
    // C/D layout: n = lane&15 ; m = v + (lane>>4)*8
    f8 accs[4] = {acc0, acc1, acc2, acc3};
    const int ncol = lane & 15, mOff = half16 * 8;
    #pragma unroll
    for (int ti = 0; ti < 4; ++ti) {
        int j = nBase + ti * 16 + ncol;
        float bj = bias[j];
        #pragma unroll
        for (int v = 0; v < 8; ++v) {
            int r = mBase + mOff + v;
            out[(long)r * DTOT + j] = fmaxf(accs[ti][v] + bj, 0.f);
        }
    }
}

extern "C" void kernel_launch(void* const* d_in, const int* in_sizes, int n_in,
                              void* d_out, int out_size, void* d_ws, size_t ws_size,
                              hipStream_t stream) {
    Params prm;
    for (int i = 0; i < 173; ++i) prm.p[i] = (i < n_in) ? (const float*)d_in[i] : nullptr;
    int B = (n_in > 1) ? in_sizes[1] / 2 : 1024;   // l_t_prev is (B,2)

    char* ws = (char*)d_ws;
    _Float16* Arow = (_Float16*)(ws);              // B x 768 f16   (1,572,864 B)
    _Float16* Bt   = (_Float16*)(ws + 1572864);    // 768 x 768 f16 (1,179,648 B)
    float*    bias = (float*)(ws + 2752512);       // 768 f32
    int*      sidx = (int*)(ws + 2755584);         // B
    int*      pos  = (int*)(ws + 2759680);         // B

    pack_b_kernel<<<(DTOT * DTOT + 255) / 256, 256, 0, stream>>>(prm, Bt, bias);
    prep_kernel<<<B, 256, 0, stream>>>(prm, sidx, Arow);
    sort_kernel<<<1, B, 0, stream>>>(sidx, pos, B);
    tower_kernel<<<B, 256, 0, stream>>>(prm, sidx, pos, Arow);
    int nBlocks = (B / 128) * 12;                  // 8 waves: 128 rows x 64 cols per block
    gemm_kernel<<<nBlocks, 256, 0, stream>>>(Arow, Bt, bias, (float*)d_out);
}
